// EnsembleRNNModule_63737314672718
// MI455X (gfx1250) — matrix operand order repaired and checked by reading, not compiled, not run on hardware
//
#include <hip/hip_runtime.h>

typedef __attribute__((ext_vector_type(16))) _Float16 v16h;
typedef __attribute__((ext_vector_type(8)))  _Float16 v8h;
typedef __attribute__((ext_vector_type(8)))  float    v8f;
typedef __attribute__((ext_vector_type(4))) unsigned int u32x4;
typedef __attribute__((ext_vector_type(8))) int         i32x8;
typedef __attribute__((ext_vector_type(4))) int         i32x4;

// Problem constants (from reference)
constexpr int W_TS = 256, NE = 16, NB = 512, NI = 10;
constexpr int PROJ = 7, IN_P = 4, EMB = 32, N_TOT = 79;
constexpr int BI = NB * NI;        // 5120
constexpr int CH = NE * BI;        // 81920 channels
constexpr float MAX_RATIO = 0.5f;
constexpr float INV_SQRT_DEG = 0.70710678118654752440f;

// ---------------------------------------------------------------------------
// TDM helper: 2D tile load Global -> LDS via Tensor Data Mover.
//   rows x EMB f32, row-major, packed into LDS at lds_off.
//   D# group0: count=1, lds_addr, global_addr, type=2.
//   D# group1: data_size=4B, tensor_dim0=EMB, tensor_dim1=rows,
//              tile_dim0=EMB, tile_dim1=rows, dim0_stride=EMB.
// Issued once per workgroup (EXEC is ignored by TDM); waited with TENSORcnt.
// This toolchain declares the 6-arg builtin (extra i32x8 group before cpol).
// ---------------------------------------------------------------------------
__device__ __forceinline__ void tdm_load_rows_f32(const float* gsrc, float* lds_dst,
                                                  unsigned rows) {
#if __has_builtin(__builtin_amdgcn_tensor_load_to_lds)
  const unsigned lds_off = (unsigned)(uintptr_t)(void*)lds_dst; // low32 = LDS byte offset
  const unsigned long long ga = (unsigned long long)(uintptr_t)gsrc;
  u32x4 g0;
  g0[0] = 1u;                                   // count=1, user mode
  g0[1] = lds_off;                              // lds_addr
  g0[2] = (unsigned)(ga & 0xFFFFFFFFu);         // global_addr[31:0]
  g0[3] = (unsigned)(ga >> 32) | 0x80000000u;   // global_addr[56:32] | type=2<<30
  i32x8 g1;
  g1[0] = (int)(2u << 16);                      // wg_mask=0, data_size=4B
  g1[1] = (int)((EMB & 0xFFFFu) << 16);         // tensor_dim0 lo16 @ [63:48]
  g1[2] = (int)((rows & 0xFFFFu) << 16);        // dim0 hi=0, tensor_dim1 lo16 @ [95:80]
  g1[3] = (int)((unsigned)EMB << 16);           // dim1 hi=0, tile_dim0 @ [127:112]
  g1[4] = (int)rows;                            // tile_dim1 @ [143:128], tile_dim2=0
  g1[5] = EMB;                                  // tensor_dim0_stride lo32
  g1[6] = 0;                                    // stride0 hi, stride1 lo
  g1[7] = 0;
  i32x4 gz4 = {0, 0, 0, 0};
  i32x8 gz8 = {0, 0, 0, 0, 0, 0, 0, 0};
  if (threadIdx.x == 0) {
    __builtin_amdgcn_tensor_load_to_lds(g0, g1, gz4, gz4, gz8, 0);
    __builtin_amdgcn_s_wait_tensorcnt(0);
  }
#else
  for (unsigned idx = threadIdx.x; idx < rows * EMB; idx += blockDim.x)
    lds_dst[idx] = gsrc[idx];
#endif
  __syncthreads();
}

// ---------------------------------------------------------------------------
// Kernel 1: h1 = gelu(emb @ hW_in^T + hb_in), f16 WMMA / f32 accumulate.
// Block = 8 waves = 8 row tiles (same e).  hW_in staged to LDS by TDM once.
// ---------------------------------------------------------------------------
__global__ __launch_bounds__(256) void k_hyper1(
    const float* __restrict__ embedding,   // (E, B, EMB)
    const float* __restrict__ hW_in,       // (E, EMB, EMB)
    const float* __restrict__ hb_in,       // (E, EMB)
    _Float16* __restrict__ h1ws)           // (E*BI, EMB) f16
{
  __shared__ float sW[EMB * EMB];          // 4 KB

  const int lane = threadIdx.x & 31;
  const int wave = (blockIdx.x * blockDim.x + threadIdx.x) >> 5;
  const int tilesPerE = BI / 16;           // 320 (divisible by 8 -> block uniform e)
  const int e = wave / tilesPerE;
  const int row0 = (wave % tilesPerE) * 16;
  const int m  = lane & 15;
  const int hi = lane >> 4;

  tdm_load_rows_f32(hW_in + (size_t)e * EMB * EMB, sW, EMB);

  // A fragment: 16x32 f16.  Lane chunks are contiguous: [hi*8,+8) and [16+hi*8,+8).
  const int bi = row0 + m;
  const float* arow = embedding + ((size_t)e * NB + (bi / NI)) * EMB;
  const float4 c0 = *(const float4*)(arow + hi * 8);
  const float4 c1 = *(const float4*)(arow + hi * 8 + 4);
  const float4 c2 = *(const float4*)(arow + 16 + hi * 8);
  const float4 c3 = *(const float4*)(arow + 16 + hi * 8 + 4);
  v16h a;
  a[0]=(_Float16)c0.x; a[1]=(_Float16)c0.y; a[2]=(_Float16)c0.z; a[3]=(_Float16)c0.w;
  a[4]=(_Float16)c1.x; a[5]=(_Float16)c1.y; a[6]=(_Float16)c1.z; a[7]=(_Float16)c1.w;
  a[8]=(_Float16)c2.x; a[9]=(_Float16)c2.y; a[10]=(_Float16)c2.z; a[11]=(_Float16)c2.w;
  a[12]=(_Float16)c3.x; a[13]=(_Float16)c3.y; a[14]=(_Float16)c3.z; a[15]=(_Float16)c3.w;

#pragma unroll
  for (int nt = 0; nt < 2; ++nt) {
    // B fragment from LDS: lane -> column o (= hW row), chunks [hi*8,+8) and [16+hi*8,+8).
    const int o = nt * 16 + m;
    const float* bcol = sW + o * EMB + hi * 8;
    v16h b;
#pragma unroll
    for (int v = 0; v < 8; ++v) {
      const int kk = (v < 4) ? 2 * v : 16 + 2 * (v - 4);
      b[2 * v]     = (_Float16)bcol[kk];
      b[2 * v + 1] = (_Float16)bcol[kk + 1];
    }
    v8f c = {};
    c = __builtin_amdgcn_wmma_f32_16x16x32_f16(false, a, false, b,
                                               (short)0, c, false, false);
#pragma unroll
    for (int r = 0; r < 8; ++r) {
      const int orow = row0 + r + hi * 8;
      float x = c[r] + hb_in[e * EMB + o];
      x = 0.5f * x * (1.0f + erff(x * 0.70710678f));   // exact gelu
      h1ws[((size_t)e * BI + orow) * EMB + o] = (_Float16)x;
    }
  }
}

// ---------------------------------------------------------------------------
// Kernel 2: off = h1 @ hW_out^T + hb_out  (79 outputs = 5 N-tiles, masked by
// value, not EXEC).  hW_out (79x32 f32) staged to LDS by TDM once per block.
// ---------------------------------------------------------------------------
__global__ __launch_bounds__(256) void k_hyper2(
    const _Float16* __restrict__ h1ws,     // (E*BI, EMB) f16
    const float* __restrict__ hW_out,      // (E, N_TOT, EMB)
    const float* __restrict__ hb_out,      // (E, N_TOT)
    float* __restrict__ offws)             // (E*BI, N_TOT)
{
  __shared__ float sW[N_TOT * EMB];        // 10112 B

  const int lane = threadIdx.x & 31;
  const int wave = (blockIdx.x * blockDim.x + threadIdx.x) >> 5;
  const int tilesPerE = BI / 16;
  const int e = wave / tilesPerE;
  const int row0 = (wave % tilesPerE) * 16;
  const int m  = lane & 15;
  const int hi = lane >> 4;

  tdm_load_rows_f32(hW_out + (size_t)e * N_TOT * EMB, sW, N_TOT);

  // A fragment (f16): two contiguous b128 chunks per lane.
  const _Float16* arow = h1ws + ((size_t)e * BI + row0 + m) * EMB;
  const v8h h0v = *(const v8h*)(arow + hi * 8);
  const v8h h1v = *(const v8h*)(arow + 16 + hi * 8);
  v16h a;
#pragma unroll
  for (int i = 0; i < 8; ++i) { a[i] = h0v[i]; a[8 + i] = h1v[i]; }

#pragma unroll
  for (int nt = 0; nt < 5; ++nt) {
    const int o = nt * 16 + m;
    const bool ok = (o < N_TOT);
    const int oc = ok ? o : (N_TOT - 1);   // clamp address, mask value
    const float* bcol = sW + oc * EMB + hi * 8;
    v16h b;
#pragma unroll
    for (int v = 0; v < 8; ++v) {
      const int kk = (v < 4) ? 2 * v : 16 + 2 * (v - 4);
      const float f0 = bcol[kk];
      const float f1 = bcol[kk + 1];
      b[2 * v]     = (_Float16)(ok ? f0 : 0.0f);
      b[2 * v + 1] = (_Float16)(ok ? f1 : 0.0f);
    }
    v8f c = {};
    c = __builtin_amdgcn_wmma_f32_16x16x32_f16(false, a, false, b,
                                               (short)0, c, false, false);
#pragma unroll
    for (int r = 0; r < 8; ++r) {
      const int orow = row0 + r + hi * 8;
      if (ok)
        offws[((size_t)e * BI + orow) * N_TOT + o] = c[r] + hb_out[e * N_TOT + o];
    }
  }
}

// ---------------------------------------------------------------------------
// Kernel 3: per-channel constrain + collapse to quadratic-form coefficients.
//   delta(z) = z^T Q z + l.z + c,  z = [x0,x1,x2,h];  h' = keep*h + delta
// ---------------------------------------------------------------------------
__global__ __launch_bounds__(256) void k_coeff(
    const float* __restrict__ off,         // (CH, N_TOT)
    const float* __restrict__ w0, const float* __restrict__ w1,
    const float* __restrict__ b0, const float* __restrict__ b1,
    const float* __restrict__ wn, const float* __restrict__ bn,
    const float* __restrict__ damping,
    float* __restrict__ coeff)             // SoA: 16 x CH
{
  const int ch = blockIdx.x * blockDim.x + threadIdx.x;
  if (ch >= CH) return;
  const int e = ch / BI;
  const float* p = off + (size_t)ch * N_TOT;

  const float* W0 = w0 + e * PROJ * IN_P;
  const float* W1 = w1 + e * PROJ * IN_P;
  const float* B0 = b0 + e * PROJ;
  const float* B1 = b1 + e * PROJ;
  const float* WN = wn + e * PROJ;

  float rn0 = 0.f, rn1 = 0.f, rb0 = 0.f, rb1 = 0.f, rwn = 0.f;
#pragma unroll
  for (int i = 0; i < PROJ * IN_P; ++i) { rn0 += W0[i] * W0[i]; rn1 += W1[i] * W1[i]; }
#pragma unroll
  for (int i = 0; i < PROJ; ++i) { rb0 += B0[i] * B0[i]; rb1 += B1[i] * B1[i]; rwn += WN[i] * WN[i]; }
  const float rbn = bn[e] * bn[e];

  float on0 = 0.f, on1 = 0.f, ob0 = 0.f, ob1 = 0.f, own = 0.f;
#pragma unroll
  for (int i = 0; i < 28; ++i) { on0 += p[i] * p[i]; on1 += p[28 + i] * p[28 + i]; }
#pragma unroll
  for (int i = 0; i < 7; ++i)  { ob0 += p[56 + i] * p[56 + i];
                                 ob1 += p[63 + i] * p[63 + i];
                                 own += p[70 + i] * p[70 + i]; }
  const float obn = p[77] * p[77];

  auto cscale = [](float rn2, float on2) {
    const float mx = fmaxf(sqrtf(rn2) * MAX_RATIO, 0.01f);
    return fminf(mx / (sqrtf(on2) + 1e-8f), 1.0f);
  };
  const float s0 = cscale(rn0, on0), s1 = cscale(rn1, on1);
  const float s2 = cscale(rb0, ob0), s3 = cscale(rb1, ob1);
  const float s4 = cscale(rwn, own), s5 = cscale(rbn, obn);

  float w0e[PROJ][IN_P], w1e[PROJ][IN_P], b0e[PROJ], b1e[PROJ], wne[PROJ];
#pragma unroll
  for (int o = 0; o < PROJ; ++o) {
#pragma unroll
    for (int i = 0; i < IN_P; ++i) {
      w0e[o][i] = W0[o * IN_P + i] + p[o * IN_P + i] * s0;
      w1e[o][i] = W1[o * IN_P + i] + p[28 + o * IN_P + i] * s1;
    }
    b0e[o] = B0[o] + p[56 + o] * s2;
    b1e[o] = B1[o] + p[63 + o] * s3;
    wne[o] = WN[o] + p[70 + o] * s4;
  }
  const float bne = bn[e] + p[77] * s5;

  float Q[IN_P][IN_P] = {}, L[IN_P] = {}, cc = 0.f;
#pragma unroll
  for (int o = 0; o < PROJ; ++o) {
    const float aw = wne[o] * INV_SQRT_DEG;
#pragma unroll
    for (int i = 0; i < IN_P; ++i) {
#pragma unroll
      for (int j = 0; j < IN_P; ++j)
        Q[i][j] = fmaf(aw * w0e[o][i], w1e[o][j], Q[i][j]);
      L[i] = fmaf(aw, b0e[o] * w1e[o][i] + b1e[o] * w0e[o][i], L[i]);
    }
    cc = fmaf(aw, b0e[o] * b1e[o], cc);
  }
  cc += bne;
  const float keep = 1.0f - 1.0f / (1.0f + expf(-damping[e]));

  coeff[0 * CH + ch] = Q[0][0];
  coeff[1 * CH + ch] = Q[1][1];
  coeff[2 * CH + ch] = Q[2][2];
  coeff[3 * CH + ch] = Q[3][3];
  coeff[4 * CH + ch] = Q[0][1] + Q[1][0];
  coeff[5 * CH + ch] = Q[0][2] + Q[2][0];
  coeff[6 * CH + ch] = Q[0][3] + Q[3][0];
  coeff[7 * CH + ch] = Q[1][2] + Q[2][1];
  coeff[8 * CH + ch] = Q[1][3] + Q[3][1];
  coeff[9 * CH + ch] = Q[2][3] + Q[3][2];
  coeff[10 * CH + ch] = L[0];
  coeff[11 * CH + ch] = L[1];
  coeff[12 * CH + ch] = L[2];
  coeff[13 * CH + ch] = L[3];
  coeff[14 * CH + ch] = cc;
  coeff[15 * CH + ch] = keep;
}

// ---------------------------------------------------------------------------
// Kernel 4: 256-step recurrence, one thread per channel.  HBM-bound:
// b96 coalesced input loads, prefetch ahead, non-temporal output stores.
// ---------------------------------------------------------------------------
struct F3 { float x, y, z; };

__global__ __launch_bounds__(256) void k_scan(
    const float* __restrict__ xs,          // (W_TS, CH, 3)
    const float* __restrict__ h0,          // (CH,)
    const float* __restrict__ coeff,       // SoA 16 x CH
    float* __restrict__ out)               // (W_TS, CH)
{
  const int ch = blockIdx.x * blockDim.x + threadIdx.x;
  if (ch >= CH) return;

  const float q00 = coeff[0 * CH + ch], q11 = coeff[1 * CH + ch];
  const float q22 = coeff[2 * CH + ch], q33 = coeff[3 * CH + ch];
  const float q01 = coeff[4 * CH + ch], q02 = coeff[5 * CH + ch];
  const float q03 = coeff[6 * CH + ch], q12 = coeff[7 * CH + ch];
  const float q13 = coeff[8 * CH + ch], q23 = coeff[9 * CH + ch];
  const float l0 = coeff[10 * CH + ch], l1 = coeff[11 * CH + ch];
  const float l2 = coeff[12 * CH + ch], l3 = coeff[13 * CH + ch];
  const float cc = coeff[14 * CH + ch], keep = coeff[15 * CH + ch];

  float h = h0[ch];
  const F3* xp = (const F3*)xs + ch;       // stride CH per timestep
  float* op = out + ch;

  for (int t = 0; t < W_TS; ++t) {
    __builtin_prefetch((const void*)(xp + (size_t)(t + 8) * CH), 0, 1);
    const F3 xv = xp[(size_t)t * CH];
    const float x0 = xv.x, x1 = xv.y, x2 = xv.z;

    float acc = cc;
    acc = fmaf(l0, x0, acc);
    acc = fmaf(l1, x1, acc);
    acc = fmaf(l2, x2, acc);
    acc = fmaf(l3, h, acc);
    acc = fmaf(q00, x0 * x0, acc);
    acc = fmaf(q11, x1 * x1, acc);
    acc = fmaf(q22, x2 * x2, acc);
    acc = fmaf(q33, h * h, acc);
    acc = fmaf(q01, x0 * x1, acc);
    acc = fmaf(q02, x0 * x2, acc);
    acc = fmaf(q03, x0 * h, acc);
    acc = fmaf(q12, x1 * x2, acc);
    acc = fmaf(q13, x1 * h, acc);
    acc = fmaf(q23, x2 * h, acc);

    h = fmaf(keep, h, acc);
    __builtin_nontemporal_store(h, op + (size_t)t * CH);
  }
}

// ---------------------------------------------------------------------------
extern "C" void kernel_launch(void* const* d_in, const int* in_sizes, int n_in,
                              void* d_out, int out_size, void* d_ws, size_t ws_size,
                              hipStream_t stream) {
  const float* inputs    = (const float*)d_in[0];
  const float* h0        = (const float*)d_in[1];
  const float* embedding = (const float*)d_in[2];
  const float* w0        = (const float*)d_in[3];
  const float* w1        = (const float*)d_in[4];
  const float* b0        = (const float*)d_in[5];
  const float* b1        = (const float*)d_in[6];
  const float* weight_n  = (const float*)d_in[7];
  const float* bias_n    = (const float*)d_in[8];
  const float* damping   = (const float*)d_in[9];
  const float* hW_in     = (const float*)d_in[10];
  const float* hb_in     = (const float*)d_in[11];
  const float* hW_out    = (const float*)d_in[12];
  const float* hb_out    = (const float*)d_in[13];
  float* out = (float*)d_out;

  char* ws = (char*)d_ws;
  _Float16* h1ws = (_Float16*)ws;                                    // CH*EMB f16
  float* offws   = (float*)(ws + (size_t)CH * EMB * sizeof(_Float16));
  float* coeff   = (float*)(ws + (size_t)CH * EMB * sizeof(_Float16)
                               + (size_t)CH * N_TOT * sizeof(float));

  const int wmmaBlocks = (CH / 16) / 8;    // 8 waves (tiles) per 256-thread block
  k_hyper1<<<wmmaBlocks, 256, 0, stream>>>(embedding, hW_in, hb_in, h1ws);
  k_hyper2<<<wmmaBlocks, 256, 0, stream>>>(h1ws, hW_out, hb_out, offws);
  k_coeff <<<CH / 256, 256, 0, stream>>>(offws, w0, w1, b0, b1,
                                         weight_n, bias_n, damping, coeff);
  k_scan  <<<CH / 256, 256, 0, stream>>>(inputs, h0, coeff, out);
}
